// GraphNet_87514253623327
// MI455X (gfx1250) — matrix operand order repaired and checked by reading, not compiled, not run on hardware
//
#include <hip/hip_runtime.h>

#define FEAT 100
#define NPAD1 112
#define KPAD1 128
#define NT1 7
#define N_NODES 20000
#define N_EDGES 320000
#define NRULES 64
#define TRULE 50
#define RULE_NUMS 16
#define NTN_OUT 200

typedef __attribute__((ext_vector_type(16))) __bf16 bf16x16;
typedef __attribute__((ext_vector_type(8)))  float  f32x8;

static __device__ __forceinline__ __bf16 f2bf(float f) {
  union { float f; unsigned u; } a; a.f = f;
  unsigned r = a.u + 0x7FFFu + ((a.u >> 16) & 1u);
  union { unsigned short s; __bf16 b; } o; o.s = (unsigned short)(r >> 16);
  return o.b;
}
static __device__ __forceinline__ float bf2f(__bf16 b) {
  union { unsigned short s; __bf16 b; } i; i.b = b;
  union { unsigned u; float f; } o; o.u = ((unsigned)i.s) << 16;
  return o.f;
}

// A-fragment (16x32 bf16, MxK): lane = h*16+m (h=lane>>4, m=lane&15)
// element e -> vgpr v=e>>1, pair p=e&1; K = (v>=4?16:0) + h*8 + (v&3)*2 + p
static __device__ __forceinline__ bf16x16 load_a_frag(const __bf16* A, int kstride,
                                                      int rowBase, int kOff, int lane) {
  int h = lane >> 4, m = lane & 15;
  const __bf16* rp = A + (size_t)(rowBase + m) * kstride + kOff;
  bf16x16 a;
#pragma unroll
  for (int e = 0; e < 16; ++e) {
    int v = e >> 1, p = e & 1;
    int k = ((v >= 4) ? 16 : 0) + h * 8 + (v & 3) * 2 + p;
    a[e] = rp[k];
  }
  return a;
}

// B-fragment (32x16 bf16, KxN) from W stored [N][kstride] row-major:
// lane = h*16+n; element e covers K = kOff + h*8 + (e<8 ? e : 8+e)
static __device__ __forceinline__ bf16x16 load_b_frag(const __bf16* W, int kstride,
                                                      int colBase, int kOff, int lane) {
  int h = lane >> 4, n = lane & 15;
  const __bf16* rp = W + (size_t)(colBase + n) * kstride + kOff + h * 8;
  bf16x16 b;
#pragma unroll
  for (int e = 0; e < 16; ++e) b[e] = rp[(e < 8) ? e : 8 + e];
  return b;
}

static __device__ __forceinline__ f32x8 wmma_bf16(bf16x16 a, bf16x16 b, f32x8 c) {
  return __builtin_amdgcn_wmma_f32_16x16x32_bf16(false, a, false, b, (short)0, c, false, false);
}

// ---------------------------------------------------------------- utilities
__global__ void k_zero(float* __restrict__ p, int n) {
  int i = blockIdx.x * blockDim.x + threadIdx.x;
  if (i < n) p[i] = 0.f;
}
__global__ void k_copy(float* __restrict__ dst, const float* __restrict__ src, int n) {
  int i = blockIdx.x * blockDim.x + threadIdx.x;
  if (i < n) dst[i] = src[i];
}
// convert f32 weight slice [n x k] (row stride srcld, col offset srcoff) to bf16 [npad x kpad]
__global__ void k_wconv(const float* __restrict__ W, int n, int k, int srcld, int srcoff,
                        __bf16* __restrict__ out, int npad, int kpad) {
  int idx = blockIdx.x * blockDim.x + threadIdx.x;
  if (idx >= npad * kpad) return;
  int r = idx / kpad, c = idx % kpad;
  float v = (r < n && c < k) ? W[r * srcld + srcoff + c] : 0.f;
  out[idx] = f2bf(v);
}

// ------------------------------------------------- generic WMMA bf16 GEMM
// C[M,N] = act(A[M,K] @ Wbf[N,kpad]^T + bias); block = 4 waves, 64 rows/block
__global__ void __launch_bounds__(128) k_gemm(
    const float* __restrict__ A, int M, int K, int lda,
    const __bf16* __restrict__ Wbf, int kpad,
    const float* __restrict__ bias,
    float* __restrict__ C, int ldc, int N, int relu_flag) {
  __shared__ __bf16 Ash[64 * KPAD1];
  int tid = threadIdx.x;
  int rowBlock = blockIdx.x * 64;
  for (int idx = tid; idx < 64 * kpad; idx += 128) {
    int r = idx / kpad, c = idx % kpad;
    int row = rowBlock + r;
    float v = (row < M && c < K) ? A[(size_t)row * lda + c] : 0.f;
    Ash[idx] = f2bf(v);
  }
  __syncthreads();
  int w = tid >> 5, lane = tid & 31;
  int rbase = w * 16;
  int nt = (N + 15) >> 4;
  int nkb = kpad >> 5;
  int n = lane & 15, hh = (lane >> 4) * 8;
  for (int ct = 0; ct < nt; ++ct) {
    f32x8 acc = {};
    for (int kb = 0; kb < nkb; ++kb) {
      bf16x16 a = load_a_frag(Ash, kpad, rbase, kb * 32, lane);
      bf16x16 b = load_b_frag(Wbf, kpad, ct * 16, kb * 32, lane);
      acc = wmma_bf16(a, b, acc);
    }
    int col = ct * 16 + n;
    float bv = (bias != nullptr && col < N) ? bias[col] : 0.f;
#pragma unroll
    for (int r = 0; r < 8; ++r) {
      int row = rowBlock + rbase + hh + r;
      if (row < M && col < N) {
        float v = acc[r] + bv;
        if (relu_flag) v = fmaxf(v, 0.f);
        C[(size_t)row * ldc + col] = v;
      }
    }
  }
}

// ---------------- fused edge kernel: m = relu(e@We^T + xWd[dst] + xWs[src] + b)
// then scatter-add into agg[dst] (segment_sum). 64 edges/block, 4 waves.
__global__ void __launch_bounds__(128) k_edge(
    const float* __restrict__ ef, const int* __restrict__ srcI, const int* __restrict__ dstI,
    const __bf16* __restrict__ WeBf, const float* __restrict__ linb,
    const float* __restrict__ xWd, const float* __restrict__ xWs,
    float* __restrict__ agg, int E) {
  __shared__ __bf16 Esh[64 * KPAD1];
  __shared__ int sIdx[64], dIdx[64];
  int tid = threadIdx.x;
  int e0 = blockIdx.x * 64;
  for (int idx = tid; idx < 64 * KPAD1; idx += 128) {
    int r = idx >> 7, c = idx & 127;
    int e = e0 + r;
    float v = (e < E && c < FEAT) ? ef[(size_t)e * FEAT + c] : 0.f;
    Esh[idx] = f2bf(v);
  }
  if (tid < 64) {
    int e = e0 + tid;
    sIdx[tid] = (e < E) ? srcI[e] : 0;
    dIdx[tid] = (e < E) ? dstI[e] : 0;
  }
  if (e0 + 64 < E)  // prefetch next tile -> global_prefetch_b8
    __builtin_prefetch(&ef[(size_t)(e0 + 64) * FEAT + tid * 50], 0, 3);
  __syncthreads();
  int w = tid >> 5, lane = tid & 31;
  int rbase = w * 16;
  f32x8 acc[NT1];
#pragma unroll
  for (int ct = 0; ct < NT1; ++ct)
#pragma unroll
    for (int r = 0; r < 8; ++r) acc[ct][r] = 0.f;
  for (int kb = 0; kb < 4; ++kb) {
    bf16x16 a = load_a_frag(Esh, KPAD1, rbase, kb * 32, lane);
#pragma unroll
    for (int ct = 0; ct < NT1; ++ct) {
      bf16x16 b = load_b_frag(WeBf, KPAD1, ct * 16, kb * 32, lane);
      acc[ct] = wmma_bf16(a, b, acc[ct]);
    }
  }
  int n = lane & 15, hh = (lane >> 4) * 8;
#pragma unroll
  for (int ct = 0; ct < NT1; ++ct) {
    int col = ct * 16 + n;
    if (col >= FEAT) continue;
    float bv = linb[col];
#pragma unroll
    for (int r = 0; r < 8; ++r) {
      int et = rbase + hh + r;
      if (e0 + et >= E) continue;
      int s = sIdx[et], d = dIdx[et];
      float m = acc[ct][r] + xWd[(size_t)d * FEAT + col] + xWs[(size_t)s * FEAT + col] + bv;
      m = fmaxf(m, 0.f);
      atomicAdd(&agg[(size_t)d * FEAT + col], m);
    }
  }
}

// ------------------------------------------------ column-sum pooling (100 cols)
__global__ void __launch_bounds__(256) k_pool(const float* __restrict__ x, int M,
                                              float* __restrict__ g) {
  __shared__ float part[FEAT];
  int tid = threadIdx.x;
  if (tid < FEAT) part[tid] = 0.f;
  __syncthreads();
  int base = blockIdx.x * 32;
  for (int idx = tid; idx < 32 * FEAT; idx += 256) {
    int r = idx / FEAT, c = idx % FEAT;
    int row = base + r;
    if (row < M) atomicAdd(&part[c], x[(size_t)row * FEAT + c]);
  }
  __syncthreads();
  if (tid < FEAT) atomicAdd(&g[tid], part[tid]);
}

// ------------- persistent sequential 2-layer tanh RNN, WMMA bf16, 16 waves
__global__ void __launch_bounds__(512) k_rnn(
    const float* __restrict__ P0,
    const __bf16* __restrict__ Whh0, const __bf16* __restrict__ Wih1,
    const __bf16* __restrict__ Whh1,
    const float* __restrict__ bhh0, const float* __restrict__ bih1,
    const float* __restrict__ bhh1,
    const float* __restrict__ ruleW, const float* __restrict__ ruleB,
    float* __restrict__ emb) {
  __shared__ __bf16 H0[64 * 256];
  __shared__ __bf16 H1[64 * 256];
  int tid = threadIdx.x;
  for (int i = tid; i < 64 * 256; i += 512) { H0[i] = f2bf(0.f); H1[i] = f2bf(0.f); }
  __syncthreads();
  int w = tid >> 5, lane = tid & 31;
  int rt = (w & 3) * 16;
  int ctb = w >> 2;                         // 4 col-tile jobs per wave
  int n = lane & 15, hh = (lane >> 4) * 8;
  for (int t = 0; t < TRULE; ++t) {
    float val[4][8];
    // layer 0: h0 = tanh(P0[t] + h0 @ Whh0^T + bhh0)   (bih0 folded into P0)
#pragma unroll
    for (int j = 0; j < 4; ++j) {
      int ct = (ctb + 4 * j) * 16;
      f32x8 acc = {};
      for (int kb = 0; kb < 8; ++kb) {
        bf16x16 a = load_a_frag(H0, 256, rt, kb * 32, lane);
        bf16x16 b = load_b_frag(Whh0, 256, ct, kb * 32, lane);
        acc = wmma_bf16(a, b, acc);
      }
      int col = ct + n;
#pragma unroll
      for (int r = 0; r < 8; ++r) {
        int bIdx = rt + hh + r;
        val[j][r] = tanhf(acc[r] + P0[(size_t)(bIdx * TRULE + t) * 256 + col] + bhh0[col]);
      }
    }
    __syncthreads();
#pragma unroll
    for (int j = 0; j < 4; ++j) {
      int ct = (ctb + 4 * j) * 16;
#pragma unroll
      for (int r = 0; r < 8; ++r) H0[(rt + hh + r) * 256 + ct + n] = f2bf(val[j][r]);
    }
    __syncthreads();
    // layer 1: h1 = tanh(h0 @ Wih1^T + h1 @ Whh1^T + bih1 + bhh1)
#pragma unroll
    for (int j = 0; j < 4; ++j) {
      int ct = (ctb + 4 * j) * 16;
      f32x8 acc = {};
      for (int kb = 0; kb < 8; ++kb) {
        bf16x16 a = load_a_frag(H0, 256, rt, kb * 32, lane);
        bf16x16 b = load_b_frag(Wih1, 256, ct, kb * 32, lane);
        acc = wmma_bf16(a, b, acc);
      }
      for (int kb = 0; kb < 8; ++kb) {
        bf16x16 a = load_a_frag(H1, 256, rt, kb * 32, lane);
        bf16x16 b = load_b_frag(Whh1, 256, ct, kb * 32, lane);
        acc = wmma_bf16(a, b, acc);
      }
      int col = ct + n;
#pragma unroll
      for (int r = 0; r < 8; ++r)
        val[j][r] = tanhf(acc[r] + bih1[col] + bhh1[col]);
    }
    __syncthreads();
#pragma unroll
    for (int j = 0; j < 4; ++j) {
      int ct = (ctb + 4 * j) * 16;
#pragma unroll
      for (int r = 0; r < 8; ++r) H1[(rt + hh + r) * 256 + ct + n] = f2bf(val[j][r]);
    }
    __syncthreads();
  }
  // rules_embedding = h1_T @ rule_lin_W^T + b   (64 x 100, tiny)
  for (int o = tid; o < NRULES * FEAT; o += 512) {
    int b = o / FEAT, nn = o % FEAT;
    float s = ruleB[nn];
    for (int k = 0; k < 256; ++k) s += bf2f(H1[b * 256 + k]) * ruleW[nn * 256 + k];
    emb[o] = s;
  }
}

// ------------------- ranker MLP + sigmoid + top-16 + gather into feat[200..]
__global__ void __launch_bounds__(256) k_topk(
    const float* __restrict__ emb,
    const float* __restrict__ W1, const float* __restrict__ b1,
    const float* __restrict__ W2, const float* __restrict__ b2,
    const float* __restrict__ W3, const float* __restrict__ b3,
    float* __restrict__ feat) {
  __shared__ float h1[64 * 128];
  __shared__ float h2[64 * 64];
  __shared__ float sc[64];
  __shared__ int idx[RULE_NUMS];
  int tid = threadIdx.x;
  for (int o = tid; o < 64 * 128; o += 256) {
    int r = o >> 7, j = o & 127;
    float s = b1[j];
    for (int k = 0; k < FEAT; ++k) s += emb[r * FEAT + k] * W1[j * FEAT + k];
    h1[o] = fmaxf(s, 0.f);
  }
  __syncthreads();
  for (int o = tid; o < 64 * 64; o += 256) {
    int r = o >> 6, j = o & 63;
    float s = b2[j];
    for (int k = 0; k < 128; ++k) s += h1[r * 128 + k] * W2[j * 128 + k];
    h2[o] = fmaxf(s, 0.f);
  }
  __syncthreads();
  if (tid < 64) {
    float s = b3[0];
    for (int k = 0; k < 64; ++k) s += h2[tid * 64 + k] * W3[k];
    sc[tid] = 1.f / (1.f + expf(-s));
  }
  __syncthreads();
  if (tid == 0) {
    for (int j = 0; j < RULE_NUMS; ++j) {
      int best = 0; float bv = -1e30f;
      for (int r = 0; r < 64; ++r)
        if (sc[r] > bv) { bv = sc[r]; best = r; }
      idx[j] = best; sc[best] = -2e30f;
    }
  }
  __syncthreads();
  for (int o = tid; o < RULE_NUMS * FEAT; o += 256) {
    int j = o / FEAT, c = o % FEAT;
    feat[NTN_OUT + o] = emb[idx[j] * FEAT + c];
  }
}

// ------------------------------- NTN + scoring head (tiny, single block)
__global__ void __launch_bounds__(256) k_head(
    const float* __restrict__ g,
    const float* __restrict__ ntnW, const float* __restrict__ ntnV,
    const float* __restrict__ ntnB, const float* __restrict__ feat,
    const float* __restrict__ fc1W, const float* __restrict__ fc1b,
    const float* __restrict__ fc2W, const float* __restrict__ fc2b,
    const float* __restrict__ fc3W, const float* __restrict__ fc3b,
    float* __restrict__ out) {
  __shared__ float g1[FEAT], g2[FEAT];
  __shared__ float fs[NTN_OUT + RULE_NUMS * FEAT];  // 1800
  __shared__ float z1[500], z2[FEAT];
  const int FTOT = NTN_OUT + RULE_NUMS * FEAT;
  int tid = threadIdx.x;
  if (tid < FEAT) { g1[tid] = g[tid]; g2[tid] = g[128 + tid]; }
  __syncthreads();
  for (int o = tid; o < RULE_NUMS * FEAT; o += 256) fs[NTN_OUT + o] = feat[NTN_OUT + o];
  for (int k = tid; k < NTN_OUT; k += 256) {
    float tp = 0.f;
    for (int i = 0; i < FEAT; ++i) {
      const float* Wp = ntnW + ((size_t)i * FEAT) * NTN_OUT + k;
      float s = 0.f;
      for (int j = 0; j < FEAT; ++j) s += Wp[(size_t)j * NTN_OUT] * g2[j];
      tp += g1[i] * s;
    }
    float lin = ntnB[k];
    for (int i = 0; i < FEAT; ++i) lin += g1[i] * ntnV[k * 2 * FEAT + i];
    for (int j = 0; j < FEAT; ++j) lin += g2[j] * ntnV[k * 2 * FEAT + FEAT + j];
    fs[k] = fmaxf(tp + lin, 0.f);
  }
  __syncthreads();
  for (int o = tid; o < 500; o += 256) {
    float s = fc1b[o];
    for (int k = 0; k < FTOT; ++k) s += fs[k] * fc1W[(size_t)o * FTOT + k];
    z1[o] = fmaxf(s, 0.f);
  }
  __syncthreads();
  for (int o = tid; o < FEAT; o += 256) {
    float s = fc2b[o];
    for (int k = 0; k < 500; ++k) s += z1[k] * fc2W[o * 500 + k];
    z2[o] = fmaxf(s, 0.f);
  }
  __syncthreads();
  if (tid == 0) {
    float s = fc3b[0];
    for (int k = 0; k < FEAT; ++k) s += z2[k] * fc3W[k];
    out[0] = 1.f / (1.f + expf(-s));
  }
}

// ============================================================== launcher
extern "C" void kernel_launch(void* const* d_in, const int* in_sizes, int n_in,
                              void* d_out, int out_size, void* d_ws, size_t ws_size,
                              hipStream_t stream) {
  (void)in_sizes; (void)n_in; (void)out_size; (void)ws_size;
  const float* nf[2] = {(const float*)d_in[0], (const float*)d_in[1]};
  const int*   ei[2] = {(const int*)d_in[2], (const int*)d_in[3]};
  const float* ef[2] = {(const float*)d_in[4], (const float*)d_in[5]};
  const float* rules = (const float*)d_in[6];
  const float* cW[2][5] = {
      {(const float*)d_in[7], (const float*)d_in[7], (const float*)d_in[7],
       (const float*)d_in[9], (const float*)d_in[11]},
      {(const float*)d_in[13], (const float*)d_in[13], (const float*)d_in[13],
       (const float*)d_in[15], (const float*)d_in[17]}};
  const float* linb[2] = {(const float*)d_in[8], (const float*)d_in[14]};
  const float* bb1[2] = {(const float*)d_in[10], (const float*)d_in[16]};
  const float* bb2[2] = {(const float*)d_in[12], (const float*)d_in[18]};
  const float* ntnW = (const float*)d_in[19];
  const float* ntnV = (const float*)d_in[20];
  const float* ntnB = (const float*)d_in[21];
  const float* r0Wih = (const float*)d_in[22]; const float* r0Whh = (const float*)d_in[23];
  const float* r0bih = (const float*)d_in[24]; const float* r0bhh = (const float*)d_in[25];
  const float* r1Wih = (const float*)d_in[26]; const float* r1Whh = (const float*)d_in[27];
  const float* r1bih = (const float*)d_in[28]; const float* r1bhh = (const float*)d_in[29];
  const float* ruleW = (const float*)d_in[30]; const float* ruleB = (const float*)d_in[31];
  const float* rkW1 = (const float*)d_in[32]; const float* rkb1 = (const float*)d_in[33];
  const float* rkW2 = (const float*)d_in[34]; const float* rkb2 = (const float*)d_in[35];
  const float* rkW3 = (const float*)d_in[36]; const float* rkb3 = (const float*)d_in[37];
  const float* fc1W = (const float*)d_in[38]; const float* fc1b = (const float*)d_in[39];
  const float* fc2W = (const float*)d_in[40]; const float* fc2b = (const float*)d_in[41];
  const float* fc3W = (const float*)d_in[42]; const float* fc3b = (const float*)d_in[43];

  char* ws = (char*)d_ws;
  size_t off = 0;
  auto alloc = [&](size_t bytes) -> void* {
    void* p = ws + off;
    off = (off + bytes + 255) & ~(size_t)255;
    return p;
  };

  __bf16* bfw[10];
  for (int i = 0; i < 10; ++i) bfw[i] = (__bf16*)alloc((size_t)NPAD1 * KPAD1 * 2);
  __bf16* Wih0b = (__bf16*)alloc(256 * 128 * 2);
  __bf16* Whh0b = (__bf16*)alloc(256 * 256 * 2);
  __bf16* Wih1b = (__bf16*)alloc(256 * 256 * 2);
  __bf16* Whh1b = (__bf16*)alloc(256 * 256 * 2);
  float* P0   = (float*)alloc((size_t)NRULES * TRULE * 256 * 4);
  float* emb  = (float*)alloc(NRULES * FEAT * 4);
  float* feat = (float*)alloc((NTN_OUT + RULE_NUMS * FEAT) * 4);
  float* gvec = (float*)alloc(256 * 4);
  float* xA   = (float*)alloc((size_t)N_NODES * FEAT * 4);
  float* xB   = (float*)alloc((size_t)N_NODES * FEAT * 4);
  float* xWd  = (float*)alloc((size_t)N_NODES * FEAT * 4);
  float* xWs  = (float*)alloc((size_t)N_NODES * FEAT * 4);
  float* agg  = (float*)alloc((size_t)N_NODES * FEAT * 4);
  float* hbuf = (float*)alloc((size_t)N_NODES * FEAT * 4);

  k_zero<<<1, 256, 0, stream>>>(gvec, 256);

  // weight conversions to bf16 [npad x kpad]
  int wg1 = (NPAD1 * KPAD1 + 255) / 256;
  for (int cv = 0; cv < 2; ++cv) {
    k_wconv<<<wg1, 256, 0, stream>>>(cW[cv][0], FEAT, FEAT, 3 * FEAT, 0,        bfw[cv * 5 + 0], NPAD1, KPAD1);
    k_wconv<<<wg1, 256, 0, stream>>>(cW[cv][1], FEAT, FEAT, 3 * FEAT, FEAT,     bfw[cv * 5 + 1], NPAD1, KPAD1);
    k_wconv<<<wg1, 256, 0, stream>>>(cW[cv][2], FEAT, FEAT, 3 * FEAT, 2 * FEAT, bfw[cv * 5 + 2], NPAD1, KPAD1);
    k_wconv<<<wg1, 256, 0, stream>>>(cW[cv][3], FEAT, FEAT, FEAT, 0, bfw[cv * 5 + 3], NPAD1, KPAD1);
    k_wconv<<<wg1, 256, 0, stream>>>(cW[cv][4], FEAT, FEAT, FEAT, 0, bfw[cv * 5 + 4], NPAD1, KPAD1);
  }
  k_wconv<<<(256 * 128 + 255) / 256, 256, 0, stream>>>(r0Wih, 256, FEAT, FEAT, 0, Wih0b, 256, 128);
  k_wconv<<<(256 * 256 + 255) / 256, 256, 0, stream>>>(r0Whh, 256, 256, 256, 0, Whh0b, 256, 256);
  k_wconv<<<(256 * 256 + 255) / 256, 256, 0, stream>>>(r1Wih, 256, 256, 256, 0, Wih1b, 256, 256);
  k_wconv<<<(256 * 256 + 255) / 256, 256, 0, stream>>>(r1Whh, 256, 256, 256, 0, Whh1b, 256, 256);

  // P0 = rules(flattened 3200x100) @ Wih0^T + bih0  (parallel precompute)
  k_gemm<<<(NRULES * TRULE + 63) / 64, 128, 0, stream>>>(
      rules, NRULES * TRULE, FEAT, FEAT, Wih0b, 128, r0bih, P0, 256, 256, 0);
  // sequential RNN (WMMA), then rules_embedding
  k_rnn<<<1, 512, 0, stream>>>(P0, Whh0b, Wih1b, Whh1b, r0bhh, r1bih, r1bhh,
                               ruleW, ruleB, emb);
  // ranker + top-k -> feat[200..1799]
  k_topk<<<1, 256, 0, stream>>>(emb, rkW1, rkb1, rkW2, rkb2, rkW3, rkb3, feat);

  // GNN passes on both graphs
  int ngrid = (N_NODES + 63) / 64;
  int cgrid = (N_NODES * FEAT + 255) / 256;
  for (int gi = 0; gi < 2; ++gi) {
    const float* xin = nf[gi];
    float* xout = xA;
    for (int cv = 0; cv < 2; ++cv) {
      const __bf16* Wd = bfw[cv * 5 + 0];
      const __bf16* We = bfw[cv * 5 + 1];
      const __bf16* Ws = bfw[cv * 5 + 2];
      const __bf16* W1 = bfw[cv * 5 + 3];
      const __bf16* W2 = bfw[cv * 5 + 4];
      k_gemm<<<ngrid, 128, 0, stream>>>(xin, N_NODES, FEAT, FEAT, Wd, KPAD1, nullptr, xWd, FEAT, FEAT, 0);
      k_gemm<<<ngrid, 128, 0, stream>>>(xin, N_NODES, FEAT, FEAT, Ws, KPAD1, nullptr, xWs, FEAT, FEAT, 0);
      k_copy<<<cgrid, 256, 0, stream>>>(agg, xin, N_NODES * FEAT);  // agg = x (residual)
      k_edge<<<N_EDGES / 64, 128, 0, stream>>>(ef[gi], ei[gi], ei[gi] + N_EDGES,
                                               We, linb[cv], xWd, xWs, agg, N_EDGES);
      k_gemm<<<ngrid, 128, 0, stream>>>(agg, N_NODES, FEAT, FEAT, W1, KPAD1, bb1[cv], hbuf, FEAT, FEAT, 1);
      k_gemm<<<ngrid, 128, 0, stream>>>(hbuf, N_NODES, FEAT, FEAT, W2, KPAD1, bb2[cv], xout, FEAT, FEAT, 1);
      xin = xout;
      xout = (xout == xA) ? xB : xA;
    }
    k_pool<<<(N_NODES + 31) / 32, 256, 0, stream>>>(xin, N_NODES, gvec + gi * 128);
  }

  // NTN + head -> scalar score
  k_head<<<1, 256, 0, stream>>>(gvec, ntnW, ntnV, ntnB, feat,
                                fc1W, fc1b, fc2W, fc2b, fc3W, fc3b, (float*)d_out);
}
